// GCC_9242769622022
// MI455X (gfx1250) — matrix operand order repaired and checked by reading, not compiled, not run on hardware
//
#include <hip/hip_runtime.h>

typedef __attribute__((ext_vector_type(16))) _Float16 v16h;
typedef __attribute__((ext_vector_type(8)))  _Float16 v8h;
typedef __attribute__((ext_vector_type(8)))  float    v8f;
typedef __attribute__((ext_vector_type(4)))  float    v4f;

#define B_    128
#define N_    12
#define K_    4096
#define F_    2049
#define FP_   2064           // freq padded to multiple of 16
#define L_    257
#define LP_   320            // lag cols padded for 4-tile blocking (5 groups * 64)
#define NP_   144            // N*N pairs (== 9 * 16)
#define TAU_  128
#define SIG_  (B_*N_)        // 1536
#define K2_   (2*FP_)        // 4128 = inner dim of lag GEMM (re/im interleaved)
#define PI2_  6.283185307179586f

// ---- workspace layout (bytes) ----
#define O_XH   0ull          // f16 [1536][4096]
#define O_B1C  12582912ull   // f16 [2064][4096]  cos basis  (col-major: [f][k])
#define O_B1S  29491200ull   // f16 [2064][4096]  -sin basis
#define O_XRE  46399488ull   // f32 [1536][2064]
#define O_XIM  59080704ull   // f32 [1536][2064]
#define O_B2   71761920ull   // f16 [320][4128]   lag basis  (col-major: [j][2f+ri])
// total ~74.4 MB

// Async global->LDS copy of 16 bytes per lane (ASYNCcnt-tracked).
__device__ __forceinline__ void async_copy16(const void* gsrc, void* ldst) {
  unsigned lds = (unsigned)(unsigned long long)ldst;   // generic addr[31:0] == LDS addr
  unsigned long long g = (unsigned long long)gsrc;
  asm volatile("global_load_async_to_lds_b128 %0, %1, off"
               :: "v"(lds), "v"(g) : "memory");
}
__device__ __forceinline__ void async_wait0() {
  asm volatile("s_wait_asynccnt 0x0" ::: "memory");
}

// ---------------- prep kernels ----------------
__global__ void k_convert_x(const float* __restrict__ x, _Float16* __restrict__ xh, int n) {
  int i = blockIdx.x * blockDim.x + threadIdx.x;
  if (i < n) xh[i] = (_Float16)x[i];
}

__global__ void k_basis1(_Float16* __restrict__ bc, _Float16* __restrict__ bs) {
  int i = blockIdx.x * blockDim.x + threadIdx.x;
  if (i >= FP_ * K_) return;
  int f = i >> 12;           // row: frequency (0..2063)
  int k = i & 4095;          // col: time sample
  int ph = (k * f) & 4095;   // exact integer phase reduction
  float ang = (float)ph * (PI2_ / 4096.0f);
  float s, c;
  __sincosf(ang, &s, &c);
  bc[i] = (_Float16)c;       // Re basis:  cos(2*pi*k*f/K)
  bs[i] = (_Float16)(-s);    // Im basis: -sin(2*pi*k*f/K)   (forward DFT)
}

__global__ void k_basis2(_Float16* __restrict__ b2) {
  int i = blockIdx.x * blockDim.x + threadIdx.x;
  if (i >= LP_ * K2_) return;
  int j  = i / K2_;          // output lag column (valid < 257, rest zero)
  int kk = i % K2_;          // 2f + ri
  int f  = kk >> 1, ri = kk & 1;
  float val = 0.0f;
  if (j < L_ && f < F_) {
    int lag = (j <= TAU_) ? j : (j + (K_ - L_));     // j or j+3839
    float w = (f == 0 || f == K_/2) ? (1.0f / K_) : (2.0f / K_);
    int ph = (f * lag) & 4095;
    float ang = (float)ph * (PI2_ / 4096.0f);
    float s, c;
    __sincosf(ang, &s, &c);
    val = ri ? (-w * s) : (w * c);   // irfft:  Gre*w*cos - Gim*w*sin
  }
  b2[i] = (_Float16)val;
}

__global__ void k_phat(float* __restrict__ Xre, float* __restrict__ Xim, int n) {
  int i = blockIdx.x * blockDim.x + threadIdx.x;
  if (i >= n) return;
  float re = Xre[i], im = Xim[i];
  float inv = 1.0f / (__builtin_sqrtf(re * re + im * im) + 1e-12f);
  Xre[i] = re * inv;
  Xim[i] = im * inv;
}

// ---------------- GEMM 1: forward DFT ----------------
// 8 waves/block: each wave one signal-tile, all share one freq-tile.
// cos/sin basis chunk async-staged to LDS once per k-step, shared by all waves.
__global__ __launch_bounds__(256) void k_gemm1(const _Float16* __restrict__ xh,
                                               const _Float16* __restrict__ bc,
                                               const _Float16* __restrict__ bs,
                                               float* __restrict__ Xre,
                                               float* __restrict__ Xim) {
  __shared__ __align__(32) _Float16 bcs[16][32];   // [local f][local k]
  __shared__ __align__(32) _Float16 bss[16][32];

  const int sg = blockIdx.x / 129;     // signal-tile group (12)
  const int ft = blockIdx.x % 129;     // freq tile (129)
  const int tid = threadIdx.x;
  const int wave = tid >> 5;
  const int l = tid & 31, lo = l & 15, hi = l >> 4;
  const int st = sg * 8 + wave;        // signal tile (96)
  const int srow = st * 16 + lo;
  const int fcol = ft * 16 + lo;

  const _Float16* ap = xh + (size_t)srow * K_;

  v8f accR = {}; v8f accI = {};
  for (int kc = 0; kc < 128; ++kc) {
    // stage basis chunk: 2 arrays * 16 cols * 32 halves = 2KB, 128 lanes * 16B
    if (tid < 128) {
      int a = tid >> 6;                // 0: cos, 1: sin
      int c = (tid & 63) >> 2;         // local column 0..15
      int s = tid & 3;                 // 16B segment 0..3
      const _Float16* src = (a ? bs : bc) + (size_t)(ft * 16 + c) * K_ + kc * 32 + s * 8;
      _Float16* dst = a ? &bss[c][s * 8] : &bcs[c][s * 8];
      async_copy16(src, dst);
    }
    async_wait0();
    __syncthreads();

    // A 16x32 f16 fragment: lanes 0-15 hold K {0..15}, lanes 16-31 K {16..31}
    int aoff = kc * 32 + hi * 16;
    v8h a0 = *(const v8h*)(ap + aoff);
    v8h a1 = *(const v8h*)(ap + aoff + 8);
    v16h a = __builtin_shufflevector(a0, a1, 0,1,2,3,4,5,6,7,8,9,10,11,12,13,14,15);
    // B 32x16 fragment from LDS: lane holds 16 consecutive K for its N column
    v16h bC = *(const v16h*)&bcs[lo][hi * 16];
    v16h bS = *(const v16h*)&bss[lo][hi * 16];
    accR = __builtin_amdgcn_wmma_f32_16x16x32_f16(false, a, false, bC, (short)0, accR, false, false);
    accI = __builtin_amdgcn_wmma_f32_16x16x32_f16(false, a, false, bS, (short)0, accI, false, false);
    __syncthreads();   // protect LDS before next stage overwrites
  }

  int r0 = st * 16 + hi * 8;
  float* pre = Xre + (size_t)r0 * FP_ + fcol;
  float* pim = Xim + (size_t)r0 * FP_ + fcol;
  #pragma unroll
  for (int r = 0; r < 8; ++r) {
    pre[(size_t)r * FP_] = accR[r];
    pim[(size_t)r * FP_] = accI[r];
  }
}

// ---------------- GEMM 3: cross-power x lag basis ----------------
// 9 waves/block (one per pair-tile), all share batch b and a group of 4 lag-tiles.
// Per 16-freq chunk: X slice (1.5KB) + lag-basis chunk (4KB) async-staged to LDS;
// cross-power A fragment built from LDS, 4 WMMAs per wave per chunk.
__global__ __launch_bounds__(288) void k_gemm3(const float* __restrict__ Xre,
                                               const float* __restrict__ Xim,
                                               const _Float16* __restrict__ b2,
                                               float* __restrict__ out) {
  __shared__ __align__(16) float xsr[12][16];      // Re X[b, sig, fchunk]
  __shared__ __align__(16) float xsi[12][16];      // Im
  __shared__ __align__(32) _Float16 bch[64][32];   // lag basis [local j][local kk]

  const int b  = blockIdx.x / 5;
  const int jg = blockIdx.x % 5;       // lag tile group of 4
  const int tid = threadIdx.x;
  const int pt = tid >> 5;             // wave == pair tile (9)
  const int l = tid & 31, lo = l & 15, hi = l >> 4;

  const int p = pt * 16 + lo;          // pair index 0..143
  const int n = p / N_, m = p % N_;

  v8f acc0 = {}, acc1 = {}, acc2 = {}, acc3 = {};
  for (int fc = 0; fc < 129; ++fc) {
    // stage lag-basis chunk: 64 cols * 32 halves = 4KB, 256 lanes * 16B
    if (tid < 256) {
      int c = tid >> 2;                // local column 0..63
      int s = tid & 3;                 // 16B segment
      const _Float16* src = b2 + (size_t)(jg * 64 + c) * K2_ + fc * 32 + s * 8;
      async_copy16(src, &bch[c][s * 8]);
    }
    // stage X slice: 2 arrays * 12 sigs * 16 freqs f32 = 1.5KB, 96 lanes * 16B
    if (tid < 96) {
      int a = tid / 48;                // 0: re, 1: im
      int r = tid % 48;
      int sig = r >> 2;
      int s = r & 3;
      const float* src = (a ? Xim : Xre) + (size_t)(b * N_ + sig) * FP_ + fc * 16 + s * 4;
      float* dst = a ? &xsi[sig][s * 4] : &xsr[sig][s * 4];
      async_copy16(src, dst);
    }
    async_wait0();
    __syncthreads();

    // Build cross-power A fragment from LDS: K = 2t + (re/im), t = hi*8 + g*4 + i
    union { v16h v; _Float16 h[16]; } A;
    #pragma unroll
    for (int g = 0; g < 2; ++g) {
      int t0 = hi * 8 + g * 4;
      v4f ar = *(const v4f*)&xsr[n][t0];
      v4f ai = *(const v4f*)&xsi[n][t0];
      v4f br = *(const v4f*)&xsr[m][t0];
      v4f bi = *(const v4f*)&xsi[m][t0];
      #pragma unroll
      for (int i = 0; i < 4; ++i) {
        float gre = ar[i] * br[i] + ai[i] * bi[i];   // Xn * conj(Xm)
        float gim = ai[i] * br[i] - ar[i] * bi[i];
        A.h[g * 8 + 2 * i]     = (_Float16)gre;
        A.h[g * 8 + 2 * i + 1] = (_Float16)gim;
      }
    }
    v16h B0 = *(const v16h*)&bch[0 * 16 + lo][hi * 16];
    v16h B1 = *(const v16h*)&bch[1 * 16 + lo][hi * 16];
    v16h B2v = *(const v16h*)&bch[2 * 16 + lo][hi * 16];
    v16h B3 = *(const v16h*)&bch[3 * 16 + lo][hi * 16];
    acc0 = __builtin_amdgcn_wmma_f32_16x16x32_f16(false, A.v, false, B0,  (short)0, acc0, false, false);
    acc1 = __builtin_amdgcn_wmma_f32_16x16x32_f16(false, A.v, false, B1,  (short)0, acc1, false, false);
    acc2 = __builtin_amdgcn_wmma_f32_16x16x32_f16(false, A.v, false, B2v, (short)0, acc2, false, false);
    acc3 = __builtin_amdgcn_wmma_f32_16x16x32_f16(false, A.v, false, B3,  (short)0, acc3, false, false);
    __syncthreads();   // protect LDS before next stage overwrites
  }

  const int p0 = pt * 16 + hi * 8;
  #pragma unroll
  for (int u = 0; u < 4; ++u) {
    int j = (jg * 4 + u) * 16 + lo;
    if (j < L_) {
      float* po = out + ((size_t)b * NP_ + p0) * L_ + j;
      v8f acc = (u == 0) ? acc0 : (u == 1) ? acc1 : (u == 2) ? acc2 : acc3;
      #pragma unroll
      for (int rr = 0; rr < 8; ++rr) po[(size_t)rr * L_] = acc[rr];
    }
  }
}

// ---------------- launcher ----------------
extern "C" void kernel_launch(void* const* d_in, const int* in_sizes, int n_in,
                              void* d_out, int out_size, void* d_ws, size_t ws_size,
                              hipStream_t stream) {
  (void)in_sizes; (void)n_in; (void)out_size; (void)ws_size;
  const float* x = (const float*)d_in[0];
  float* out = (float*)d_out;
  char* ws = (char*)d_ws;

  _Float16* xh  = (_Float16*)(ws + O_XH);
  _Float16* b1c = (_Float16*)(ws + O_B1C);
  _Float16* b1s = (_Float16*)(ws + O_B1S);
  float*    Xre = (float*)   (ws + O_XRE);
  float*    Xim = (float*)   (ws + O_XIM);
  _Float16* b2  = (_Float16*)(ws + O_B2);

  int nx = SIG_ * K_;
  k_convert_x<<<(nx + 255) / 256, 256, 0, stream>>>(x, xh, nx);

  int nb1 = FP_ * K_;
  k_basis1<<<(nb1 + 255) / 256, 256, 0, stream>>>(b1c, b1s);

  int nb2 = LP_ * K2_;
  k_basis2<<<(nb2 + 255) / 256, 256, 0, stream>>>(b2);

  k_gemm1<<<12 * 129, 256, 0, stream>>>(xh, b1c, b1s, Xre, Xim);

  int nph = SIG_ * FP_;
  k_phat<<<(nph + 255) / 256, 256, 0, stream>>>(Xre, Xim, nph);

  k_gemm3<<<B_ * 5, 288, 0, stream>>>(Xre, Xim, b2, out);
}
